// BjorckLinear_82815559401622
// MI455X (gfx1250) — compile-verified, matrix operand order checked by reading, not run on hardware
//
#include <hip/hip_runtime.h>

// ---------------------------------------------------------------------------
// BjorckLinear for MI455X (gfx1250, wave32, WMMA).
//
// out = X @ bjorck(W)^T,  X:[131072,512] f32, W:[512,512] f32
// bjorck: 10x  W <- 1.5*W - 0.5 * W @ (W^T @ W)
//
// Two paths:
//  * Bjorck iterations (20 small 512^3 GEMMs, L2-resident, precision-
//    critical): generic fp32 WMMA kernel (V_WMMA_F32_16X16X4_F32) with
//    stride-parameterized operands so W^T W needs no transpose pass.
//  * Main GEMM (131072x512x512, compute-bound): split-precision f16 WMMA
//    (V_WMMA_F32_16X16X32_F16). Each f32 = hi(f16) + lo(f16); the product
//    uses hi*hi + hi*lo + lo*hi with f32 accumulate -> ~2^-22 relative
//    error, on par with fp32, at the f16 matrix pipe's rate. Wave tile is
//    16x64 (4 subtiles) so each A hi/lo fragment feeds 4 B columns:
//    20 b128 ds loads per 12 WMMAs per K-chunk.
// ---------------------------------------------------------------------------

typedef __attribute__((ext_vector_type(2)))  float    v2f;
typedef __attribute__((ext_vector_type(8)))  float    v8f;
typedef __attribute__((ext_vector_type(4)))  float    f32x4;
typedef __attribute__((ext_vector_type(4)))  _Float16 v4h;
typedef __attribute__((ext_vector_type(8)))  _Float16 v8h;
typedef __attribute__((ext_vector_type(16))) _Float16 v16h;

#define TILE_M 64
#define TILE_N 64
#define KC     32
#define LDA_S  33    // fp32 kernel: padded LDS stride for A tile
#define LDB_S  65    // fp32 kernel: padded LDS stride for B tile
#define LDH    40    // f16 kernel: halves per LDS row (80 B: 16B-aligned,
                     // r*20 mod 64 distinct for 16 rows -> conflict-free b128)
#define TN2    128   // f16 kernel: N tile (wave tile 16x64)

// ============================ fp32 WMMA GEMM ===============================
// D[M,N] = alpha * (A @ B) + beta * Caux   (Caux optional, same layout as D)
// elemA(r,c) = A[r*rsA + c*csA], elemB(r,c) = B[r*rsB + c*csB]
__global__ __launch_bounds__(256)
void bjorck_gemm_f32_wmma(const float* __restrict__ A, long rsA, long csA,
                          const float* __restrict__ B, long rsB, long csB,
                          const float* __restrict__ Caux, float alpha, float beta,
                          float* __restrict__ D, long ldd, int K)
{
    __shared__ float As[TILE_M * LDA_S];
    __shared__ float Bs[KC * LDB_S];

    const int tid  = threadIdx.x;
    const int lane = tid & 31;
    const int wave = tid >> 5;
    const int lm   = lane & 15;
    const int half = lane >> 4;

    const long tile_m = (long)blockIdx.x * TILE_M;
    const long tile_n = (long)blockIdx.y * TILE_N;

    const int rowg = (wave >> 1) * 16;
    const int colg = (wave & 1) * 32;

    v8f acc0 = {};
    v8f acc1 = {};

    for (int k0 = 0; k0 < K; k0 += KC) {
        #pragma unroll
        for (int idx = tid; idx < TILE_M * KC; idx += 256) {
            int r = idx >> 5;
            int c = idx & (KC - 1);
            As[r * LDA_S + c] = A[(tile_m + r) * rsA + (long)(k0 + c) * csA];
        }
        #pragma unroll
        for (int idx = tid; idx < KC * TILE_N; idx += 256) {
            int r = idx >> 6;
            int c = idx & (TILE_N - 1);
            Bs[r * LDB_S + c] = B[(long)(k0 + r) * rsB + (tile_n + c) * csB];
        }
        __syncthreads();

        #pragma unroll
        for (int kk = 0; kk < KC; kk += 4) {
            const int ka = kk + 2 * half;
            v2f a;
            a.x = As[(rowg + lm) * LDA_S + ka];
            a.y = As[(rowg + lm) * LDA_S + ka + 1];
            v2f b0, b1;
            b0.x = Bs[ka * LDB_S + colg + lm];
            b0.y = Bs[(ka + 1) * LDB_S + colg + lm];
            b1.x = Bs[ka * LDB_S + colg + 16 + lm];
            b1.y = Bs[(ka + 1) * LDB_S + colg + 16 + lm];

            acc0 = __builtin_amdgcn_wmma_f32_16x16x4_f32(
                false, a, false, b0, (short)0, acc0, false, false);
            acc1 = __builtin_amdgcn_wmma_f32_16x16x4_f32(
                false, a, false, b1, (short)0, acc1, false, false);
        }
        __syncthreads();
    }

    #pragma unroll
    for (int r = 0; r < 8; ++r) {
        const long row = tile_m + rowg + r + 8 * half;
        const long c0  = tile_n + colg + lm;
        const long c1  = c0 + 16;
        float v0 = alpha * acc0[r];
        float v1 = alpha * acc1[r];
        if (Caux != nullptr) {
            v0 += beta * Caux[row * ldd + c0];
            v1 += beta * Caux[row * ldd + c1];
        }
        D[row * ldd + c0] = v0;
        D[row * ldd + c1] = v1;
    }
}

// ===================== split-precision f16 WMMA GEMM =======================
// out[b,o] = sum_k X[b,k] * W[o,k];  X:[batch,K] f32, W:[N,K] f32, K=lda=512.
// 64 (M) x 128 (N) block tile, 8 waves; each wave owns a 16x64 strip
// (4 WMMA subtiles) so A fragments are reused 4x.
__global__ __launch_bounds__(256)
void xwt_gemm_f16split_wmma(const float* __restrict__ X,
                            const float* __restrict__ W,
                            float* __restrict__ out,
                            int K, long lda)
{
    __shared__ _Float16 Ah[TILE_M * LDH];
    __shared__ _Float16 Al[TILE_M * LDH];
    __shared__ _Float16 Bh[TN2 * LDH];
    __shared__ _Float16 Bl[TN2 * LDH];

    const int tid  = threadIdx.x;
    const int lane = tid & 31;
    const int wave = tid >> 5;
    const int lm   = lane & 15;
    const int half = lane >> 4;

    const long tile_m = (long)blockIdx.x * TILE_M;
    const long tile_n = (long)blockIdx.y * TN2;

    const int rowg = (wave >> 1) * 16;   // 4 row groups of 16
    const int colg = (wave & 1) * 64;    // 2 col groups of 64

    v8f acc[4] = {{}, {}, {}, {}};

    for (int k0 = 0; k0 < K; k0 += KC) {
        // ---- stage + split A: 64 rows x 32 k (512 float4, 2 per thread)
        #pragma unroll
        for (int i = 0; i < 2; ++i) {
            const int fid = tid + 256 * i;
            const int r   = fid >> 3;           // 8 float4 per 32-float row
            const int c4  = (fid & 7) * 4;
            f32x4 x = *(const f32x4*)&X[(tile_m + r) * lda + k0 + c4];
            v4h hi, lo;
            #pragma unroll
            for (int j = 0; j < 4; ++j) {
                _Float16 h = (_Float16)x[j];
                hi[j] = h;
                lo[j] = (_Float16)(x[j] - (float)h);
            }
            *(v4h*)&Ah[r * LDH + c4] = hi;
            *(v4h*)&Al[r * LDH + c4] = lo;
            if (k0 + KC < K)
                __builtin_prefetch(&X[(tile_m + r) * lda + k0 + KC + c4], 0, 0);
        }
        // ---- stage + split B: 128 rows x 32 k (1024 float4, 4 per thread)
        #pragma unroll
        for (int i = 0; i < 4; ++i) {
            const int fid = tid + 256 * i;
            const int r   = fid >> 3;
            const int c4  = (fid & 7) * 4;
            f32x4 w = *(const f32x4*)&W[(tile_n + r) * lda + k0 + c4];
            v4h hi, lo;
            #pragma unroll
            for (int j = 0; j < 4; ++j) {
                _Float16 h = (_Float16)w[j];
                hi[j] = h;
                lo[j] = (_Float16)(w[j] - (float)h);
            }
            *(v4h*)&Bh[r * LDH + c4] = hi;
            *(v4h*)&Bl[r * LDH + c4] = lo;
            if (k0 + KC < K)
                __builtin_prefetch(&W[(tile_n + r) * lda + k0 + KC + c4], 0, 0);
        }
        __syncthreads();

        // ---- A fragments: two aligned b128 runs per 16x32 operand tile
        // (ISA 7.12.2: lane half h holds K in [8h,8h+8) u [16+8h,16+8h+8))
        const int aoff = (rowg + lm) * LDH + 8 * half;
        v8h ah0 = *(const v8h*)&Ah[aoff];
        v8h ah1 = *(const v8h*)&Ah[aoff + 16];
        v8h al0 = *(const v8h*)&Al[aoff];
        v8h al1 = *(const v8h*)&Al[aoff + 16];
        v16h aH = __builtin_shufflevector(ah0, ah1, 0,1,2,3,4,5,6,7,8,9,10,11,12,13,14,15);
        v16h aL = __builtin_shufflevector(al0, al1, 0,1,2,3,4,5,6,7,8,9,10,11,12,13,14,15);

        // ---- 4 B column subtiles; interleaved accumulate chains
        #pragma unroll
        for (int j = 0; j < 4; ++j) {
            const int boff = (colg + 16 * j + lm) * LDH + 8 * half;
            v8h bh0 = *(const v8h*)&Bh[boff];
            v8h bh1 = *(const v8h*)&Bh[boff + 16];
            v8h bl0 = *(const v8h*)&Bl[boff];
            v8h bl1 = *(const v8h*)&Bl[boff + 16];
            v16h bH = __builtin_shufflevector(bh0, bh1, 0,1,2,3,4,5,6,7,8,9,10,11,12,13,14,15);
            v16h bL = __builtin_shufflevector(bl0, bl1, 0,1,2,3,4,5,6,7,8,9,10,11,12,13,14,15);

            acc[j] = __builtin_amdgcn_wmma_f32_16x16x32_f16(
                false, aH, false, bH, (short)0, acc[j], false, false);
            acc[j] = __builtin_amdgcn_wmma_f32_16x16x32_f16(
                false, aH, false, bL, (short)0, acc[j], false, false);
            acc[j] = __builtin_amdgcn_wmma_f32_16x16x32_f16(
                false, aL, false, bH, (short)0, acc[j], false, false);
        }
        __syncthreads();
    }

    // ---- epilogue: C/D layout (VGPR r -> M = r + 8*half, N = lm)
    #pragma unroll
    for (int j = 0; j < 4; ++j) {
        #pragma unroll
        for (int r = 0; r < 8; ++r) {
            const long row = tile_m + rowg + r + 8 * half;
            const long col = tile_n + colg + 16 * j + lm;
            out[row * lda + col] = acc[j][r];
        }
    }
}

// ================================ launch ===================================
extern "C" void kernel_launch(void* const* d_in, const int* in_sizes, int n_in,
                              void* d_out, int out_size, void* d_ws, size_t ws_size,
                              hipStream_t stream) {
    (void)n_in; (void)out_size; (void)ws_size;

    constexpr long N = 512;   // D_IN == D_OUT == 512
    const float* X = (const float*)d_in[0];   // [batch, 512]
    const float* W = (const float*)d_in[1];   // [512, 512]
    float* out = (float*)d_out;

    const long batch = (long)in_sizes[0] / N;

    // Workspace: two weight ping-pong buffers + Gram matrix (3 MB total).
    float* W0 = (float*)d_ws;
    float* W1 = W0 + N * N;
    float* G  = W1 + N * N;

    hipMemcpyAsync(W0, W, N * N * sizeof(float), hipMemcpyDeviceToDevice, stream);

    dim3 blk(256);
    dim3 gridS(N / TILE_M, N / TILE_N);       // 8 x 8 blocks for 512x512

    for (int it = 0; it < 10; ++it) {
        // G = W0^T @ W0 : elemA(i,m) = W0[m*512 + i] -> rsA=1, csA=512
        bjorck_gemm_f32_wmma<<<gridS, blk, 0, stream>>>(
            W0, 1, N,  W0, N, 1,  nullptr, 1.0f, 0.0f,  G, N, (int)N);
        // W1 = 1.5*W0 - 0.5*(W0 @ G)
        bjorck_gemm_f32_wmma<<<gridS, blk, 0, stream>>>(
            W0, N, 1,  G, N, 1,  W0, -0.5f, 1.5f,  W1, N, (int)N);
        float* t = W0; W0 = W1; W1 = t;
    }

    // out = X @ W0^T via split-precision f16 WMMA (compute-bound path).
    dim3 gridC((unsigned)(batch / TILE_M), (unsigned)(N / TN2));
    xwt_gemm_f16split_wmma<<<gridC, blk, 0, stream>>>(X, W0, out, (int)N, N);
}
